// Attention_60258391163572
// MI455X (gfx1250) — matrix operand order repaired and checked by reading, not compiled, not run on hardware
//
#include <hip/hip_runtime.h>
#include <cstddef>

// ---------------------------------------------------------------------------
// Types
// ---------------------------------------------------------------------------
typedef __bf16 bf16;
typedef __attribute__((ext_vector_type(16))) __bf16 bf16x16;
typedef __attribute__((ext_vector_type(8)))  __bf16 bf16x8;
typedef __attribute__((ext_vector_type(8)))  float  f32x8;
typedef __attribute__((ext_vector_type(4)))  unsigned int u32x4;
typedef __attribute__((ext_vector_type(8)))  int          i32x8;
typedef __attribute__((ext_vector_type(4)))  int          i32x4;

#define DEVINL __device__ __forceinline__

DEVINL bf16x16 cat8(bf16x8 lo, bf16x8 hi) {
    return __builtin_shufflevector(lo, hi, 0,1,2,3,4,5,6,7,8,9,10,11,12,13,14,15);
}

// Low 32 bits of a generic pointer to LDS = group-segment byte offset.
DEVINL unsigned lds_off(const void* p) {
    return (unsigned)(unsigned long long)p;
}

// A fragment (16x32 bf16, M x K). Source row-major [16][ldk] at `base`.
// Lane L: row M = L%16; VGPR0-3 = K in [8*(L>=16), +8), VGPR4-7 = +16.
DEVINL bf16x16 load_a_frag(const bf16* base, int ldk, int lane) {
    int m  = lane & 15;
    int ko = (lane >> 4) * 8;
    const bf16* p = base + m * ldk + ko;
    bf16x8 lo = *(const bf16x8*)(p);
    bf16x8 hi = *(const bf16x8*)(p + 16);
    return cat8(lo, hi);
}

// B fragment (32x16 bf16, K x N). Source stored K-contiguous per column n,
// rows [n][k] with leading dim ldk. Lane L: col N = L%16; VGPR0-3 = K in [8*(L>=16), +8), VGPR4-7 = +16.
DEVINL bf16x16 load_b_frag(const bf16* base, int ldk, int lane) {
    int n  = lane & 15;
    int ko = (lane >> 4) * 8;
    const bf16* p = base + n * ldk + ko;
    bf16x8 lo = *(const bf16x8*)(p);
    bf16x8 hi = *(const bf16x8*)(p + 16);
    return cat8(lo, hi);
}

DEVINL f32x8 wmma_bf16(bf16x16 a, bf16x16 b, f32x8 c) {
    return __builtin_amdgcn_wmma_f32_16x16x32_bf16(false, a, false, b, (short)0, c,
                                                   false, false);
}

DEVINL float rmax16(float x) {
    #pragma unroll
    for (int m = 1; m <= 8; m <<= 1) x = fmaxf(x, __shfl_xor(x, m, 32));
    return x;
}
DEVINL float rsum16(float x) {
    #pragma unroll
    for (int m = 1; m <= 8; m <<= 1) x += __shfl_xor(x, m, 32);
    return x;
}

// Async global->LDS copy of 32 bytes per lane (two b128 ops; the instruction
// offset is added to BOTH the LDS and global address per the ISA pseudocode).
DEVINL void async_copy_32B(unsigned ldsDst, unsigned gOff, unsigned long long gBase) {
    asm volatile("global_load_async_to_lds_b128 %0, %1, %2\n\t"
                 "global_load_async_to_lds_b128 %0, %1, %2 offset:16"
                 :: "v"(ldsDst), "v"(gOff), "s"(gBase) : "memory");
}
DEVINL void async_copy_16B(unsigned ldsDst, unsigned gOff, unsigned long long gBase) {
    asm volatile("global_load_async_to_lds_b128 %0, %1, %2"
                 :: "v"(ldsDst), "v"(gOff), "s"(gBase) : "memory");
}
DEVINL void wait_async0() {
    asm volatile("s_wait_asynccnt 0x0" ::: "memory");
}

// ---------------------------------------------------------------------------
// Problem constants
// ---------------------------------------------------------------------------
static constexpr int BATCH   = 2;
static constexpr int SEQ     = 2048;
static constexpr int HIDDEN  = 4096;
static constexpr int NHEADS  = 32;
static constexpr int HDIM    = 128;
static constexpr int ROWS    = BATCH * SEQ;   // 4096
static constexpr int N3H     = 3 * HIDDEN;    // 12288

// ---------------------------------------------------------------------------
// 1) fp32 -> bf16 conversion
// ---------------------------------------------------------------------------
__global__ __launch_bounds__(256) void cvt_f32_bf16(const float* __restrict__ in,
                                                    bf16* __restrict__ out, int n) {
    int i = blockIdx.x * 256 + threadIdx.x;
    if (i < n) out[i] = (bf16)in[i];
}

// ---------------------------------------------------------------------------
// 2/5) Tiled WMMA GEMM:  C[M,N] = A[M,K] * B[N,K]^T   (A,B bf16 row-major)
//     Block 128(M) x 64(N), k-step 32, 8 waves; tiles staged with
//     global_load_async_to_lds_b128 (ASYNCcnt).
// ---------------------------------------------------------------------------
template <bool OUT_F32>
__global__ __launch_bounds__(256) void gemm_bt(const bf16* __restrict__ A,
                                               const bf16* __restrict__ B,
                                               void* __restrict__ Cout,
                                               int M, int N, int K) {
    __shared__ __align__(16) bf16 As[128][32];
    __shared__ __align__(16) bf16 Bs[64][32];

    const int tid  = threadIdx.x;
    const int lane = tid & 31;
    const int wave = tid >> 5;
    const int wm   = wave >> 1;
    const int wn   = wave & 1;
    const int m0   = blockIdx.y * 128;
    const int n0   = blockIdx.x * 64;

    f32x8 acc[2][2] = {};

    const int ar = tid >> 1, ah = tid & 1;   // A tile: row 0..127, 32B half
    const int br = tid >> 2, bq = tid & 3;   // B tile: row 0..63, 16B quarter

    const unsigned long long Abase = (unsigned long long)A;
    const unsigned long long Bbase = (unsigned long long)B;
    const unsigned aDst = lds_off(&As[ar][ah * 16]);
    const unsigned bDst = lds_off(&Bs[br][bq * 8]);

    for (int k0 = 0; k0 < K; k0 += 32) {
        // Async-stage A [128][32] and B [64][32] tiles directly into LDS.
        async_copy_32B(aDst, (unsigned)(((size_t)(m0 + ar) * K + k0 + ah * 16)
                                        * sizeof(bf16)), Abase);
        async_copy_16B(bDst, (unsigned)(((size_t)(n0 + br) * K + k0 + bq * 8)
                                        * sizeof(bf16)), Bbase);
        if (k0 + 32 < K) {
            __builtin_prefetch(A + (size_t)(m0 + ar) * K + k0 + 32, 0, 1);
            __builtin_prefetch(B + (size_t)(n0 + br) * K + k0 + 32, 0, 1);
        }
        wait_async0();
        __syncthreads();

        bf16x16 bfrag[2];
        #pragma unroll
        for (int ni = 0; ni < 2; ++ni)
            bfrag[ni] = load_b_frag(&Bs[wn * 32 + ni * 16][0], 32, lane);
        #pragma unroll
        for (int mi = 0; mi < 2; ++mi) {
            bf16x16 afrag = load_a_frag(&As[wm * 32 + mi * 16][0], 32, lane);
            #pragma unroll
            for (int ni = 0; ni < 2; ++ni)
                acc[mi][ni] = wmma_bf16(afrag, bfrag[ni], acc[mi][ni]);
        }
        __syncthreads();
    }

    #pragma unroll
    for (int mi = 0; mi < 2; ++mi) {
        #pragma unroll
        for (int ni = 0; ni < 2; ++ni) {
            #pragma unroll
            for (int v = 0; v < 8; ++v) {
                int row = m0 + wm * 32 + mi * 16 + (lane >> 4) * 8 + v;
                int col = n0 + wn * 32 + ni * 16 + (lane & 15);
                if (OUT_F32)
                    ((float*)Cout)[(size_t)row * N + col] = acc[mi][ni][v];
                else
                    ((bf16*)Cout)[(size_t)row * N + col] = (bf16)acc[mi][ni][v];
            }
        }
    }
}

// ---------------------------------------------------------------------------
// 3) RoPE + head split:  proj[B,S,3H] -> q/k (rotated), v in [B*H, S, D] bf16
// ---------------------------------------------------------------------------
__global__ __launch_bounds__(256) void rope_split(const bf16* __restrict__ proj,
                                                  bf16* __restrict__ q,
                                                  bf16* __restrict__ k,
                                                  bf16* __restrict__ v) {
    int idx = blockIdx.x * 256 + threadIdx.x;
    int d   = idx & 63;
    int h   = (idx >> 6) & (NHEADS - 1);
    int bs  = idx >> 11;
    int s   = bs & (SEQ - 1);
    int b   = bs >> 11;

    float ang = (float)s * __expf(-(float)(2 * d) * (9.210340371976184f / 128.0f));
    float sn, cs;
    __sincosf(ang, &sn, &cs);

    size_t base = (size_t)bs * N3H + h * HDIM;
    float q1 = (float)proj[base + d],          q2 = (float)proj[base + d + 64];
    float k1 = (float)proj[base + HIDDEN + d], k2 = (float)proj[base + HIDDEN + d + 64];

    size_t o = (((size_t)b * NHEADS + h) * SEQ + s) * HDIM + d;
    q[o]      = (bf16)(q1 * cs - q2 * sn);
    q[o + 64] = (bf16)(q2 * cs + q1 * sn);
    k[o]      = (bf16)(k1 * cs - k2 * sn);
    k[o + 64] = (bf16)(k2 * cs + k1 * sn);
    v[o]      = proj[base + 2 * HIDDEN + d];
    v[o + 64] = proj[base + 2 * HIDDEN + d + 64];
}

// ---------------------------------------------------------------------------
// 4) Flash attention (causal). Grid: (SEQ/64, B*NHEADS). Block: 128 (4 waves).
//    K tile is DMA'd by the Tensor Data Mover; V tile is transposed manually.
// ---------------------------------------------------------------------------
__global__ __launch_bounds__(128) void flash_attn(const bf16* __restrict__ Q,
                                                  const bf16* __restrict__ Kr,
                                                  const bf16* __restrict__ V,
                                                  bf16* __restrict__ Out) {
    __shared__ __align__(16) bf16 Ks[32][128];   // [key][d]   (TDM destination)
    __shared__ __align__(16) bf16 Vs[128][32];   // [d][key]   (transposed)
    __shared__ __align__(16) bf16 Ps[4][16][32]; // per-wave P tile

    const int tid  = threadIdx.x;
    const int lane = tid & 31;
    const int wave = tid >> 5;
    const int bh   = blockIdx.y;
    const int h    = bh & (NHEADS - 1);
    const int b    = bh >> 5;
    const int q0   = blockIdx.x * 64 + wave * 16;

    const bf16* Qb = Q + ((size_t)bh * SEQ + q0) * HDIM;
    bf16x16 qf[4];
    {
        int m = lane & 15, ko = (lane >> 4) * 8;
        #pragma unroll
        for (int c = 0; c < 4; ++c) {
            const bf16* p = Qb + m * HDIM + c * 32 + ko;
            qf[c] = cat8(*(const bf16x8*)p, *(const bf16x8*)(p + 16));
        }
    }

    f32x8 o[8] = {};
    float mrow[8], lrow[8];
    #pragma unroll
    for (int v = 0; v < 8; ++v) { mrow[v] = -3.0e38f; lrow[v] = 0.0f; }

    const float scale  = 0.08838834764831845f;   // 1/sqrt(128)
    const int   qrbase = q0 + 8 * (lane >> 4);
    const int   kend   = blockIdx.x * 64 + 64;
    const int   cr = tid >> 2, cq = tid & 3;
    const unsigned ksBase = lds_off(&Ks[0][0]);

    for (int kk = 0; kk < kend; kk += 32) {
        // ---- K tile [32][128] via Tensor Data Mover (one wave issues) ----
        if (wave == 0) {
            unsigned long long ga =
                (unsigned long long)(Kr + ((size_t)bh * SEQ + kk) * HDIM);
            u32x4 g0;
            g0[0] = 1u;                                   // count=1, user desc
            g0[1] = ksBase;                               // lds_addr
            g0[2] = (unsigned)ga;                         // global_addr[31:0]
            g0[3] = (unsigned)(ga >> 32) | 0x80000000u;   // addr[56:32] | type=2
            i32x8 g1;
            g1[0] = 0x00010000;          // data_size=1 (2 bytes)
            g1[1] = (HDIM << 16);        // tensor_dim0 = 128
            g1[2] = (int)(2048u << 16);  // tensor_dim1 = SEQ
            g1[3] = (HDIM << 16);        // tile_dim0 = 128
            g1[4] = 32;                  // tile_dim1 = 32
            g1[5] = HDIM;                // tensor_dim0_stride = 128
            g1[6] = 0;
            g1[7] = 0;
            i32x4 gz = {0, 0, 0, 0};
            asm volatile("tensor_load_to_lds %0, %1, %2, %3"
                         :: "s"(g0), "s"(g1), "s"(gz), "s"(gz) : "memory");
            __builtin_amdgcn_s_wait_tensorcnt(0);
        }
        // ---- V tile transposed: Vs[d][key] ----
        {
            const bf16* g = V + ((size_t)bh * SEQ + kk + cr) * HDIM + cq * 32;
            bf16x8 t[4];
            #pragma unroll
            for (int i = 0; i < 4; ++i) t[i] = *(const bf16x8*)(g + i * 8);
            #pragma unroll
            for (int j = 0; j < 32; ++j) Vs[cq * 32 + j][cr] = t[j >> 3][j & 7];
        }
        __syncthreads();

        // ---- scores ----
        f32x8 s0 = {}, s1 = {};
        #pragma unroll
        for (int c = 0; c < 4; ++c) {
            bf16x16 kb0 = load_b_frag(&Ks[0][0]  + c * 32, 128, lane);
            bf16x16 kb1 = load_b_frag(&Ks[16][0] + c * 32, 128, lane);
            s0 = wmma_bf16(qf[c], kb0, s0);
            s1 = wmma_bf16(qf[c], kb1, s1);
        }

        // ---- mask + scale + online softmax ----
        int col0 = kk + (lane & 15);
        #pragma unroll
        for (int v = 0; v < 8; ++v) {
            int qr = qrbase + v;
            float e0 = s0[v] * scale; if (col0 > qr)      e0 = -1.0e30f;
            float e1 = s1[v] * scale; if (col0 + 16 > qr) e1 = -1.0e30f;
            float rmx  = rmax16(fmaxf(e0, e1));
            float mnew = fmaxf(mrow[v], rmx);
            float corr = __expf(mrow[v] - mnew);
            mrow[v] = mnew;
            float p0 = __expf(e0 - mnew);
            float p1 = __expf(e1 - mnew);
            lrow[v] = lrow[v] * corr + rsum16(p0 + p1);
            #pragma unroll
            for (int j = 0; j < 8; ++j) o[j][v] *= corr;
            int prow = 8 * (lane >> 4) + v;
            Ps[wave][prow][lane & 15]        = (bf16)p0;
            Ps[wave][prow][(lane & 15) + 16] = (bf16)p1;
        }
        __syncthreads();

        // ---- O += P(16x32) @ V(32x128) ----
        bf16x16 pf = load_a_frag(&Ps[wave][0][0], 32, lane);
        #pragma unroll
        for (int j = 0; j < 8; ++j) {
            bf16x16 vb = load_b_frag(&Vs[j * 16][0], 32, lane);
            o[j] = wmma_bf16(pf, vb, o[j]);
        }
        __syncthreads();
    }

    #pragma unroll
    for (int v = 0; v < 8; ++v) {
        float inv_l = 1.0f / lrow[v];
        int row = qrbase + v;
        #pragma unroll
        for (int j = 0; j < 8; ++j) {
            size_t oi = ((size_t)b * SEQ + row) * HIDDEN + h * HDIM + j * 16 + (lane & 15);
            Out[oi] = (bf16)(o[j][v] * inv_l);
        }
    }
}

// ---------------------------------------------------------------------------
// Host-side orchestration
// ---------------------------------------------------------------------------
extern "C" void kernel_launch(void* const* d_in, const int* in_sizes, int n_in,
                              void* d_out, int out_size, void* d_ws, size_t ws_size,
                              hipStream_t stream) {
    (void)in_sizes; (void)n_in; (void)out_size; (void)ws_size;

    const float* hs = (const float*)d_in[0];
    const float* Wp = (const float*)d_in[3];
    const float* Wo = (const float*)d_in[4];
    float*       out = (float*)d_out;

    char* w = (char*)d_ws;
    bf16* hsb  = (bf16*)(w);
    bf16* wpb  = (bf16*)(w + (size_t)33554432);
    bf16* wob  = (bf16*)(w + (size_t)134217728);
    bf16* proj = (bf16*)(w + (size_t)167772160);
    bf16* qro  = (bf16*)(w + (size_t)33554432);    // overlay wpb (dead after GEMM)
    bf16* kro  = (bf16*)(w + (size_t)67108864);
    bf16* vbf  = (bf16*)(w + (size_t)100663296);
    bf16* aout = (bf16*)(w);                       // overlay hsb (dead after GEMM)

    const int nHS = ROWS * HIDDEN;
    const int nWP = N3H * HIDDEN;
    const int nWO = HIDDEN * HIDDEN;

    cvt_f32_bf16<<<nHS / 256, 256, 0, stream>>>(hs, hsb, nHS);
    cvt_f32_bf16<<<nWP / 256, 256, 0, stream>>>(Wp, wpb, nWP);
    cvt_f32_bf16<<<nWO / 256, 256, 0, stream>>>(Wo, wob, nWO);

    gemm_bt<false><<<dim3(N3H / 64, ROWS / 128), 256, 0, stream>>>(
        hsb, wpb, proj, ROWS, N3H, HIDDEN);

    const int nR = ROWS * NHEADS * 64;
    rope_split<<<nR / 256, 256, 0, stream>>>(proj, qro, kro, vbf);

    flash_attn<<<dim3(SEQ / 64, BATCH * NHEADS), 128, 0, stream>>>(qro, kro, vbf, aout);

    gemm_bt<true><<<dim3(HIDDEN / 64, ROWS / 128), 256, 0, stream>>>(
        aout, wob, out, ROWS, HIDDEN, HIDDEN);
}
